// Head_46600395161875
// MI455X (gfx1250) — matrix operand order repaired and checked by reading, not compiled, not run on hardware
//
#include <hip/hip_runtime.h>
#include <stdint.h>

typedef _Float16 f16;
typedef __attribute__((ext_vector_type(16))) _Float16 v16h;
typedef __attribute__((ext_vector_type(8)))  _Float16 v8h;
typedef __attribute__((ext_vector_type(8)))  float    v8f;
typedef __attribute__((ext_vector_type(4)))  float    v4f;

#define D_MODEL 512
#define HEAD    64
#define SEQ     4096
#define BATCH   4

// 1/sqrt(HEAD) * log2(e): softmax computed in exp2 domain (v_exp_f32 is exp2)
#define Q_SCALE (0.125f * 1.4426950408889634f)

__device__ __forceinline__ float fast_exp2(float x) {
#if __has_builtin(__builtin_amdgcn_exp2f)
  return __builtin_amdgcn_exp2f(x);
#else
  return exp2f(x);
#endif
}

__device__ __forceinline__ v8f wmma16(v16h a, v16h b, v8f c) {
  // D = A(16x32 f16) * B(32x16 f16) + C(16x16 f32)
  return __builtin_amdgcn_wmma_f32_16x16x32_f16(false, a, false, b, (short)0, c,
                                                false, false);
}

__device__ __forceinline__ v16h cat8(v8h lo, v8h hi) {
  return __builtin_shufflevector(lo, hi, 0, 1, 2, 3, 4, 5, 6, 7, 8, 9, 10, 11,
                                 12, 13, 14, 15);
}

// Async global->LDS copy of 16 bytes (GLOBAL_LOAD_ASYNC_TO_LDS_B128, GV mode).
__device__ __forceinline__ void async_copy_b128(uint32_t ldsAddr,
                                                unsigned long long gaddr) {
  asm volatile("global_load_async_to_lds_b128 %0, %1, off"
               :: "v"(ldsAddr), "v"(gaddr)
               : "memory");
}

__device__ __forceinline__ void wait_asynccnt0() {
  asm volatile("s_wait_asynccnt 0x0" ::: "memory");
}

// ---------------------------------------------------------------------------
// Phase 1: Q,K,V = x @ W{q,k,v}, output f16, Q_SCALE folded into Q.
// Block = 256 threads = 8 waves. Wave w owns rows [blk*128 + w*16, +16).
// W chunk (32x64) staged transposed into LDS as f16: Wt[mat][h][k].
// ---------------------------------------------------------------------------
__global__ __launch_bounds__(256) void qkv_proj_kernel(
    const float* __restrict__ x, const float* __restrict__ Wq,
    const float* __restrict__ Wk, const float* __restrict__ Wv,
    f16* __restrict__ qo, f16* __restrict__ ko, f16* __restrict__ vo) {
  __shared__ __align__(16) f16 Wt[3][HEAD][32];  // 12 KB

  const int tid  = threadIdx.x;
  const int lane = tid & 31;
  const int wave = tid >> 5;
  const int half = lane >> 4;   // 0: lanes 0-15, 1: lanes 16-31
  const int mn   = lane & 15;
  const int rowBase = blockIdx.x * 128 + wave * 16;  // flattened [B*T] row

  const float* Wsrc[3] = {Wq, Wk, Wv};

  v8f acc[3][4];
#pragma unroll
  for (int m = 0; m < 3; ++m)
#pragma unroll
    for (int nt = 0; nt < 4; ++nt)
      acc[m][nt] = (v8f){0.f, 0.f, 0.f, 0.f, 0.f, 0.f, 0.f, 0.f};

  const int b0 = half * 8;  // A-fragment K-group base within 32-chunk

  for (int kc = 0; kc < D_MODEL; kc += 32) {
    __syncthreads();
    // ---- stage transposed W chunk: each thread moves 8 f16 per matrix ----
    {
      const int idx = tid * 8;       // 256*8 = 2048 = 32*64
      const int kk  = idx >> 6;      // 0..31
      const int c0  = idx & 63;      // multiple of 8
#pragma unroll
      for (int m = 0; m < 3; ++m) {
        const float* src = Wsrc[m] + (size_t)(kc + kk) * HEAD + c0;
        v4f f0 = *(const v4f*)src;
        v4f f1 = *(const v4f*)(src + 4);
#pragma unroll
        for (int i = 0; i < 4; ++i) {
          Wt[m][c0 + i][kk]     = (f16)f0[i];
          Wt[m][c0 + 4 + i][kk] = (f16)f1[i];
        }
      }
    }
    __syncthreads();

    // ---- A fragment from x (f32 -> f16), per CDNA5 16-bit A layout ----
    const float* xrow = x + (size_t)(rowBase + mn) * D_MODEL + kc;
    v4f g0 = *(const v4f*)(xrow + b0);
    v4f g1 = *(const v4f*)(xrow + b0 + 4);
    v4f g2 = *(const v4f*)(xrow + 16 + b0);
    v4f g3 = *(const v4f*)(xrow + 16 + b0 + 4);
    v16h a;
#pragma unroll
    for (int i = 0; i < 4; ++i) {
      a[i]      = (f16)g0[i];
      a[4 + i]  = (f16)g1[i];
      a[8 + i]  = (f16)g2[i];
      a[12 + i] = (f16)g3[i];
    }

    // ---- B fragments from LDS (8 f16 at k=8*half, 8 at k=16+8*half) + WMMA ----
#pragma unroll
    for (int m = 0; m < 3; ++m) {
#pragma unroll
      for (int nt = 0; nt < 4; ++nt) {
        const f16* wp = &Wt[m][nt * 16 + mn][half * 8];
        v16h bf = cat8(*(const v8h*)wp, *(const v8h*)(wp + 16));
        acc[m][nt] = wmma16(a, bf, acc[m][nt]);
      }
    }
  }

  // ---- store D tiles (row = r + 8*half, col = nt*16 + mn) ----
  f16* outs[3] = {qo, ko, vo};
  const float scl[3] = {Q_SCALE, 1.0f, 1.0f};
#pragma unroll
  for (int m = 0; m < 3; ++m) {
#pragma unroll
    for (int nt = 0; nt < 4; ++nt) {
#pragma unroll
      for (int r = 0; r < 8; ++r) {
        const int row = rowBase + r + 8 * half;
        outs[m][(size_t)row * HEAD + nt * 16 + mn] = (f16)(acc[m][nt][r] * scl[m]);
      }
    }
  }
}

// ---------------------------------------------------------------------------
// Phase 2: flash attention with diagonal-zero mask, exp2-domain softmax.
// Block = 4 waves (128 threads); wave owns one 16-row query tile; block
// marches over keys in chunks of 64 with double-buffered LDS tiles:
//   K staged via async global->LDS copies (ASYNCcnt), V transposed on store.
// ---------------------------------------------------------------------------
#define P2_WAVES 4
#define KCHUNK   64

__global__ __launch_bounds__(128) void attn_kernel(
    const f16* __restrict__ qf, const f16* __restrict__ kf,
    const f16* __restrict__ vf, float* __restrict__ out) {
  __shared__ __align__(16) f16 Klds[2][KCHUNK][HEAD];     // [buf][key][h] 16 KB
  __shared__ __align__(16) f16 Vt[2][HEAD][KCHUNK];       // [buf][h][key] 16 KB
  __shared__ __align__(16) f16 Plds[P2_WAVES][16][KCHUNK];// per-wave P     8 KB

  const int tid  = threadIdx.x;
  const int lane = tid & 31;
  const int wave = tid >> 5;
  const int half = lane >> 4;
  const int mn   = lane & 15;
  const int b0   = half * 8;

  const int b    = blockIdx.x >> 6;  // 64 blocks per batch
  const int qRow = ((blockIdx.x & 63) * P2_WAVES + wave) * 16;  // within batch
  const size_t batchOff = (size_t)b * SEQ * HEAD;

  // stager: thread moves 64 bytes of K (async) and 64 bytes of V (transpose)
  const int srow = tid >> 1;         // 0..63
  const int scol = (tid & 1) * 32;   // 0 or 32 (f16 units)
  auto stage = [&](int buf, int s0) {
    const f16* ksrc = kf + batchOff + (size_t)(s0 + srow) * HEAD + scol;
    uint32_t kdst = (uint32_t)(uintptr_t)&Klds[buf][srow][scol];
#pragma unroll
    for (int i = 0; i < 4; ++i)
      async_copy_b128(kdst + 16 * i,
                      (unsigned long long)(uintptr_t)ksrc + 16 * i);

    const f16* vsrc = vf + batchOff + (size_t)(s0 + srow) * HEAD + scol;
    if (s0 + KCHUNK < SEQ) __builtin_prefetch(vsrc + KCHUNK * HEAD, 0, 1);
#pragma unroll
    for (int j = 0; j < 4; ++j) {
      v8h vv = *(const v8h*)(vsrc + 8 * j);
#pragma unroll
      for (int i = 0; i < 8; ++i) Vt[buf][scol + 8 * j + i][srow] = vv[i];
    }
  };

  // ---- loop-invariant Q fragments (scale already folded in) ----
  const f16* qrow = qf + batchOff + (size_t)(qRow + mn) * HEAD;
  v16h aq0 = cat8(*(const v8h*)(qrow + half * 16), *(const v8h*)(qrow + half * 16 + 8));
  v16h aq1 = cat8(*(const v8h*)(qrow + 32 + half * 16), *(const v8h*)(qrow + 40 + half * 16));

  v8f O[4];
#pragma unroll
  for (int ht = 0; ht < 4; ++ht)
    O[ht] = (v8f){0.f, 0.f, 0.f, 0.f, 0.f, 0.f, 0.f, 0.f};
  float mrow[8], lrow[8];
#pragma unroll
  for (int r = 0; r < 8; ++r) { mrow[r] = -1e30f; lrow[r] = 0.f; }

  // prologue: fill buffer 0
  stage(0, 0);
  wait_asynccnt0();
  __syncthreads();

  for (int s0 = 0; s0 < SEQ; s0 += KCHUNK) {
    const int cur = (s0 >> 6) & 1;
    // issue next chunk's staging; overlaps with this chunk's WMMAs
    if (s0 + KCHUNK < SEQ) stage(cur ^ 1, s0 + KCHUNK);

    // ---- S = Q K^T for four 16-key subtiles ----
    v8f S[4];
#pragma unroll
    for (int t = 0; t < 4; ++t) {
      const f16* kp0 = &Klds[cur][t * 16 + mn][half * 16];
      const f16* kp1 = &Klds[cur][t * 16 + mn][32 + half * 16];
      v16h bk0 = cat8(*(const v8h*)kp0, *(const v8h*)(kp0 + 8));
      v16h bk1 = cat8(*(const v8h*)kp1, *(const v8h*)(kp1 + 8));
      v8f s = (v8f){0.f, 0.f, 0.f, 0.f, 0.f, 0.f, 0.f, 0.f};
      s = wmma16(aq0, bk0, s);
      s = wmma16(aq1, bk1, s);
      S[t] = s;
    }

    // ---- diagonal mask: wei[i][i] = 0 (S'=0 in exp2 domain too) ----
#pragma unroll
    for (int t = 0; t < 4; ++t) {
#pragma unroll
      for (int r = 0; r < 8; ++r) {
        if (qRow + r + 8 * half == s0 + t * 16 + mn) S[t][r] = 0.f;
      }
    }

    // ---- online softmax (row stats per-reg, reduced over 16-lane halves) ----
#pragma unroll
    for (int r = 0; r < 8; ++r) {
      float v = fmaxf(fmaxf(S[0][r], S[1][r]), fmaxf(S[2][r], S[3][r]));
      v = fmaxf(v, __shfl_xor(v, 1, 32));
      v = fmaxf(v, __shfl_xor(v, 2, 32));
      v = fmaxf(v, __shfl_xor(v, 4, 32));
      v = fmaxf(v, __shfl_xor(v, 8, 32));
      const float mnew = fmaxf(mrow[r], v);
      const float fac  = fast_exp2(mrow[r] - mnew);
      mrow[r] = mnew;
      float p[4];
      float rs = 0.f;
#pragma unroll
      for (int t = 0; t < 4; ++t) {
        p[t] = fast_exp2(S[t][r] - mnew);
        rs += p[t];
      }
      rs += __shfl_xor(rs, 1, 32);
      rs += __shfl_xor(rs, 2, 32);
      rs += __shfl_xor(rs, 4, 32);
      rs += __shfl_xor(rs, 8, 32);
      lrow[r] = lrow[r] * fac + rs;
#pragma unroll
      for (int ht = 0; ht < 4; ++ht) O[ht][r] *= fac;
      const int m = r + 8 * half;
#pragma unroll
      for (int t = 0; t < 4; ++t) Plds[wave][m][t * 16 + mn] = (f16)p[t];
    }

    // ---- O += P V  (A = P from wave-private LDS, B = V transposed) ----
    const f16* pp = &Plds[wave][mn][0];
    v16h ap0 = cat8(*(const v8h*)(pp + half * 16), *(const v8h*)(pp + half * 16 + 8));
    v16h ap1 = cat8(*(const v8h*)(pp + 32 + half * 16), *(const v8h*)(pp + 40 + half * 16));
#pragma unroll
    for (int ht = 0; ht < 4; ++ht) {
      const f16* vp0 = &Vt[cur][ht * 16 + mn][half * 16];
      const f16* vp1 = &Vt[cur][ht * 16 + mn][32 + half * 16];
      v16h bv0 = cat8(*(const v8h*)vp0, *(const v8h*)(vp0 + 8));
      v16h bv1 = cat8(*(const v8h*)vp1, *(const v8h*)(vp1 + 8));
      O[ht] = wmma16(ap0, bv0, O[ht]);
      O[ht] = wmma16(ap1, bv1, O[ht]);
    }

    // drain own async copies for the next buffer, then block-wide sync:
    // after this barrier everyone's next tile is resident and all reads of
    // the current tile are done (protects next iteration's overwrite).
    wait_asynccnt0();
    __syncthreads();
  }

  // ---- epilogue: out = O / l ----
#pragma unroll
  for (int ht = 0; ht < 4; ++ht) {
#pragma unroll
    for (int r = 0; r < 8; ++r) {
      const int m = r + 8 * half;
      out[batchOff + (size_t)(qRow + m) * HEAD + ht * 16 + mn] =
          O[ht][r] / lrow[r];
    }
  }
}

// ---------------------------------------------------------------------------
extern "C" void kernel_launch(void* const* d_in, const int* in_sizes, int n_in,
                              void* d_out, int out_size, void* d_ws,
                              size_t ws_size, hipStream_t stream) {
  const float* x  = (const float*)d_in[0];
  const float* Wq = (const float*)d_in[1];
  const float* Wk = (const float*)d_in[2];
  const float* Wv = (const float*)d_in[3];

  const size_t nProj = (size_t)BATCH * SEQ * HEAD;  // 1M elements
  f16* qf = (f16*)d_ws;
  f16* kf = qf + nProj;
  f16* vf = kf + nProj;

  // Phase 1: 16384 rows / (8 waves * 16 rows) = 128 blocks
  qkv_proj_kernel<<<128, 256, 0, stream>>>(x, Wq, Wk, Wv, qf, kf, vf);
  // Phase 2: 4 batches * 64 blocks (each: 4 waves * 16 query rows)
  attn_kernel<<<256, 128, 0, stream>>>(qf, kf, vf, (float*)d_out);
}
